// CrossAttention_20409684591358
// MI455X (gfx1250) — matrix operand order repaired and checked by reading, not compiled, not run on hardware
//
#include <hip/hip_runtime.h>

#define BATCH 8
#define H     8
#define DK    128
#define DV    128
#define S1    2048
#define S2    2048
#define D1    1024

typedef _Float16 half_t;
typedef __attribute__((ext_vector_type(4)))  _Float16 v4h;
typedef __attribute__((ext_vector_type(8)))  _Float16 v8h;
typedef __attribute__((ext_vector_type(16))) _Float16 v16h;
typedef __attribute__((ext_vector_type(8)))  float    v8f;
typedef int i32x4 __attribute__((vector_size(16)));
typedef __attribute__((address_space(1))) i32x4 glb_i32x4;
typedef __attribute__((address_space(3))) i32x4 lds_i32x4;

#if __has_builtin(__builtin_amdgcn_global_load_async_to_lds_b128)
#define HAS_ASYNC_LDS 1
#else
#define HAS_ASYNC_LDS 0
#endif

// 16-byte global -> LDS copy; async (ASYNCcnt-tracked) when available.
__device__ __forceinline__ void copy16(half_t* ldst, const half_t* gsrc) {
#if HAS_ASYNC_LDS
  __builtin_amdgcn_global_load_async_to_lds_b128(
      (glb_i32x4*)gsrc, (lds_i32x4*)ldst, 0, 0);
#else
  *(v8h*)ldst = *(const v8h*)gsrc;
#endif
}
__device__ __forceinline__ void wait_async() {
#if HAS_ASYNC_LDS
#if __has_builtin(__builtin_amdgcn_s_wait_asynccnt)
  __builtin_amdgcn_s_wait_asynccnt(0);
#else
  asm volatile("s_wait_asynccnt 0" ::: "memory");
#endif
#endif
}

// ---- WMMA fragment loaders (CDNA5 16-bit layouts, wave32) ----
// A 16x32 f16: lane m = lane&15, half = lane>>4
//   v0..3 hold K = half*8 + {0..7}, v4..7 hold K = 16 + half*8 + {0..7}
__device__ __forceinline__ v16h frag_a(const half_t* base, int hs) {
  union { v16h v; v8h h[2]; } u;
  u.h[0] = *(const v8h*)(base + hs * 8);
  u.h[1] = *(const v8h*)(base + 16 + hs * 8);
  return u.v;
}
// B 32x16 f16: lane n = lane&15, half = lane>>4; v0..3 hold K = half*8 + {0..7}, v4..7 hold K = 16 + half*8 + {0..7}
__device__ __forceinline__ v16h frag_b(const half_t* base, int hs) {
  union { v16h v; v8h h[2]; } u;
  u.h[0] = *(const v8h*)(base + hs * 8);
  u.h[1] = *(const v8h*)(base + 16 + hs * 8);
  return u.v;
}
__device__ __forceinline__ v8f wmma16(v16h a, v16h b, v8f c) {
  return __builtin_amdgcn_wmma_f32_16x16x32_f16(false, a, false, b, (short)0, c,
                                                false, false);
}

// =====================================================================
// Kernel 1: C[16384x1024] = A(f32) @ W(f32), f16 output scattered into
// head-split layouts.  mode 0: q [B][H][S1][DK]
//                      mode 1: k [B][H][S2][DK]
//                      mode 2: v transposed [B][H][DV][S2]
// Tile 128x128x32, 256 threads = 8 waves (4x2).  Global tile loads are
// register-double-buffered so they overlap the WMMA chain.
// =====================================================================
__global__ __launch_bounds__(256) void proj_kernel(
    const float* __restrict__ A, const float* __restrict__ W,
    half_t* __restrict__ out, int mode) {
  __shared__ half_t As[128 * 40];   // [m][k], padded stride
  __shared__ half_t Bs[128 * 40];   // [n][k] (transposed), padded stride
  const int t = threadIdx.x;
  const int lane = t & 31, wave = t >> 5;
  const int ln = lane & 15, hs = lane >> 4;
  const int wm = wave & 3, wn = wave >> 2;
  const int m0 = blockIdx.y * 128, n0 = blockIdx.x * 128;

  v8f zero = {0.f,0.f,0.f,0.f,0.f,0.f,0.f,0.f};
  v8f acc[2][4];
#pragma unroll
  for (int i = 0; i < 2; ++i)
#pragma unroll
    for (int j = 0; j < 4; ++j) acc[i][j] = zero;

  const int ar = t >> 3, ac = (t & 7) * 4;     // A staging: float4 per thread
  const int bc = (t & 31) * 4, br = t >> 5;    // B staging: float4, transpose

  float4 fa[4], fb[4];
  auto fetch = [&](int k0) {
#pragma unroll
    for (int p = 0; p < 4; ++p)
      fa[p] = *(const float4*)(A + (size_t)(m0 + ar + p * 32) * D1 + k0 + ac);
#pragma unroll
    for (int p = 0; p < 4; ++p)
      fb[p] = *(const float4*)(W + (size_t)(k0 + br + p * 8) * 1024 + n0 + bc);
  };
  fetch(0);

  for (int k0 = 0; k0 < D1; k0 += 32) {
#pragma unroll
    for (int p = 0; p < 4; ++p) {
      v4h hv;
      hv[0] = (half_t)fa[p].x; hv[1] = (half_t)fa[p].y;
      hv[2] = (half_t)fa[p].z; hv[3] = (half_t)fa[p].w;
      *(v4h*)&As[(ar + p * 32) * 40 + ac] = hv;
    }
#pragma unroll
    for (int p = 0; p < 4; ++p) {
      int kk = br + p * 8;
      Bs[(bc + 0) * 40 + kk] = (half_t)fb[p].x;
      Bs[(bc + 1) * 40 + kk] = (half_t)fb[p].y;
      Bs[(bc + 2) * 40 + kk] = (half_t)fb[p].z;
      Bs[(bc + 3) * 40 + kk] = (half_t)fb[p].w;
    }
    __syncthreads();
    if (k0 + 32 < D1) fetch(k0 + 32);   // overlap next tile with WMMAs

    v16h a0 = frag_a(&As[(wm * 32 + ln) * 40], hs);
    v16h a1 = frag_a(&As[(wm * 32 + 16 + ln) * 40], hs);
    v16h bbv[4];
#pragma unroll
    for (int tn = 0; tn < 4; ++tn)
      bbv[tn] = frag_b(&Bs[(wn * 64 + tn * 16 + ln) * 40], hs);
#pragma unroll
    for (int tn = 0; tn < 4; ++tn) {
      acc[0][tn] = wmma16(a0, bbv[tn], acc[0][tn]);
      acc[1][tn] = wmma16(a1, bbv[tn], acc[1][tn]);
    }
    __syncthreads();
  }

#pragma unroll
  for (int tm = 0; tm < 2; ++tm)
#pragma unroll
    for (int tn = 0; tn < 4; ++tn) {
      int n = n0 + wn * 64 + tn * 16 + ln;
      int hh = n >> 7, d = n & 127;
#pragma unroll
      for (int r = 0; r < 8; ++r) {
        int row = m0 + wm * 32 + tm * 16 + r + 8 * hs;
        int bb = row >> 11;            // S=2048
        int s  = row & 2047;
        half_t val = (half_t)acc[tm][tn][r];
        size_t bh = (size_t)bb * H + hh;
        if (mode == 0)      out[(bh * S1 + s) * DK + d] = val;
        else if (mode == 1) out[(bh * S2 + s) * DK + d] = val;
        else                out[(bh * DV + d) * S2 + s] = val;  // transposed V
      }
    }
}

// =====================================================================
// Kernel 2: flash attention.  Block = 128 threads = 4 waves, 64 q rows.
// Q held in registers (4 A-frags / wave).  64-key KV tiles are streamed
// into double-buffered LDS with GLOBAL_LOAD_ASYNC_TO_LDS_B128 so the next
// tile's fill overlaps the current tile's 32 WMMAs + softmax.
// =====================================================================
__global__ __launch_bounds__(128) void attn_kernel(
    const half_t* __restrict__ q, const half_t* __restrict__ k,
    const half_t* __restrict__ vt, half_t* __restrict__ aout) {
  __shared__ half_t Ks[2][64 * 136];      // [key][dk], padded
  __shared__ half_t VTs[2][128 * 72];     // [dv][key], padded
  __shared__ half_t Ps[4 * 16 * 72];      // per-wave P tile [qrow][key]

  const int t = threadIdx.x;
  const int lane = t & 31, wave = t >> 5;
  const int ln = lane & 15, hs = lane >> 4;
  const int qb = blockIdx.x, h = blockIdx.y, b = blockIdx.z;

  const size_t bh = (size_t)b * H + h;
  const half_t* qrow = q + (bh * S1 + (size_t)qb * 64 + wave * 16 + ln) * DK;
  const half_t* kbase = k + bh * (size_t)S2 * DK;
  const half_t* vbase = vt + bh * (size_t)DV * S2;

  // per-thread staging coordinates (8 x 16B chunks for each tile)
  const int krow = t >> 4, kcol = (t & 15) * 8;   // + p*8 rows
  const int vrow = t >> 3, vcol = (t & 7) * 8;    // + p*16 rows

  auto stage = [&](int kb, int buf) {
#pragma unroll
    for (int p = 0; p < 8; ++p) {
      int row = krow + p * 8;
      copy16(&Ks[buf][row * 136 + kcol],
             kbase + (size_t)(kb + row) * DK + kcol);
    }
#pragma unroll
    for (int p = 0; p < 8; ++p) {
      int dv = vrow + p * 16;
      copy16(&VTs[buf][dv * 72 + vcol],
             vbase + (size_t)dv * S2 + kb + vcol);
    }
  };

  v16h qf[4];
#pragma unroll
  for (int c = 0; c < 4; ++c) qf[c] = frag_a(qrow + c * 32, hs);

  v8f zero = {0.f,0.f,0.f,0.f,0.f,0.f,0.f,0.f};
  v8f o[8];
#pragma unroll
  for (int i = 0; i < 8; ++i) o[i] = zero;
  float mrow[8], lrow[8];
#pragma unroll
  for (int r = 0; r < 8; ++r) { mrow[r] = -3.0e38f; lrow[r] = 0.f; }

  const float scale = 0.08838834764831845f;  // 1/sqrt(128)

  stage(0, 0);
  int cur = 0;
  for (int kb = 0; kb < S2; kb += 64, cur ^= 1) {
    wait_async();        // s_wait_asynccnt 0: this wave's tile fill done
    __syncthreads();     // all waves' fills visible
    if (kb + 64 < S2) stage(kb + 64, cur ^ 1);   // overlap next fill

    // S = Q @ K^T  (4 n-tiles x 4 dk-chunks = 16 WMMA)
    v8f s4[4];
#pragma unroll
    for (int tn = 0; tn < 4; ++tn) {
      v16h bf[4];
#pragma unroll
      for (int c = 0; c < 4; ++c)
        bf[c] = frag_b(&Ks[cur][(tn * 16 + ln) * 136 + c * 32], hs);
      v8f acc = zero;
#pragma unroll
      for (int c = 0; c < 4; ++c) acc = wmma16(qf[c], bf[c], acc);
#pragma unroll
      for (int r = 0; r < 8; ++r) acc[r] *= scale;
      s4[tn] = acc;
    }

    // online softmax; rows live at r + 8*hs, cols across the 16-lane group
    half_t* prow = &Ps[wave * (16 * 72)];
#pragma unroll
    for (int r = 0; r < 8; ++r) {
      float mx = fmaxf(fmaxf(s4[0][r], s4[1][r]), fmaxf(s4[2][r], s4[3][r]));
#pragma unroll
      for (int off = 1; off < 16; off <<= 1)
        mx = fmaxf(mx, __shfl_xor(mx, off, 32));
      float nm = fmaxf(mrow[r], mx);
      float alpha = __expf(mrow[r] - nm);
      mrow[r] = nm;
      float rs = 0.f;
#pragma unroll
      for (int tn = 0; tn < 4; ++tn) {
        float p = __expf(s4[tn][r] - nm);
        rs += p;
        prow[(r + 8 * hs) * 72 + tn * 16 + ln] = (half_t)p;
      }
#pragma unroll
      for (int off = 1; off < 16; off <<= 1) rs += __shfl_xor(rs, off, 32);
      lrow[r] = lrow[r] * alpha + rs;
#pragma unroll
      for (int td = 0; td < 8; ++td) o[td][r] *= alpha;
    }
    __syncthreads();  // P visible in LDS before A-fragment reload

    // O += P @ V   (2 key-chunks x 8 dv-tiles = 16 WMMA)
    const half_t* pb = prow + ln * 72;
#pragma unroll
    for (int c2 = 0; c2 < 2; ++c2) {
      v16h pf = frag_a(pb + c2 * 32, hs);
#pragma unroll
      for (int g = 0; g < 2; ++g) {     // groups of 4 preloaded B-frags
        v16h bf[4];
#pragma unroll
        for (int j = 0; j < 4; ++j)
          bf[j] = frag_b(&VTs[cur][((g * 4 + j) * 16 + ln) * 72 + c2 * 32], hs);
#pragma unroll
        for (int j = 0; j < 4; ++j)
          o[g * 4 + j] = wmma16(pf, bf[j], o[g * 4 + j]);
      }
    }
  }

  // epilogue: O / l -> f16, merged-head layout [B*S1][H*DV]
  const size_t orow = (size_t)b * S1 + (size_t)qb * 64 + wave * 16;
#pragma unroll
  for (int r = 0; r < 8; ++r) {
    float inv = 1.0f / lrow[r];
    size_t rbase = (orow + r + 8 * hs) * (size_t)(H * DV) + (size_t)h * DV;
#pragma unroll
    for (int td = 0; td < 8; ++td)
      aout[rbase + td * 16 + ln] = (half_t)(o[td][r] * inv);
  }
}

// =====================================================================
// Kernel 3: out = attn(f16)[16384x1024] @ Wo(f32) + bo -> f32
// =====================================================================
__global__ __launch_bounds__(256) void outproj_kernel(
    const half_t* __restrict__ A, const float* __restrict__ W,
    const float* __restrict__ bias, float* __restrict__ out) {
  __shared__ half_t As[128 * 40];
  __shared__ half_t Bs[128 * 40];
  const int t = threadIdx.x;
  const int lane = t & 31, wave = t >> 5;
  const int ln = lane & 15, hs = lane >> 4;
  const int wm = wave & 3, wn = wave >> 2;
  const int m0 = blockIdx.y * 128, n0 = blockIdx.x * 128;

  v8f zero = {0.f,0.f,0.f,0.f,0.f,0.f,0.f,0.f};
  v8f acc[2][4];
#pragma unroll
  for (int i = 0; i < 2; ++i)
#pragma unroll
    for (int j = 0; j < 4; ++j) acc[i][j] = zero;

  const int arow = t >> 2, acol = (t & 3) * 8;  // A: v8h per thread, 2 passes
  const int bc = (t & 31) * 4, br = t >> 5;

  v8h ga[2];
  float4 fb[4];
  auto fetch = [&](int k0) {
#pragma unroll
    for (int p = 0; p < 2; ++p)
      ga[p] = *(const v8h*)(A + (size_t)(m0 + arow + p * 64) * 1024 + k0 + acol);
#pragma unroll
    for (int p = 0; p < 4; ++p)
      fb[p] = *(const float4*)(W + (size_t)(k0 + br + p * 8) * 1024 + n0 + bc);
  };
  fetch(0);

  for (int k0 = 0; k0 < 1024; k0 += 32) {
#pragma unroll
    for (int p = 0; p < 2; ++p)
      *(v8h*)&As[(arow + p * 64) * 40 + acol] = ga[p];
#pragma unroll
    for (int p = 0; p < 4; ++p) {
      int kk = br + p * 8;
      Bs[(bc + 0) * 40 + kk] = (half_t)fb[p].x;
      Bs[(bc + 1) * 40 + kk] = (half_t)fb[p].y;
      Bs[(bc + 2) * 40 + kk] = (half_t)fb[p].z;
      Bs[(bc + 3) * 40 + kk] = (half_t)fb[p].w;
    }
    __syncthreads();
    if (k0 + 32 < 1024) fetch(k0 + 32);

    v16h a0 = frag_a(&As[(wm * 32 + ln) * 40], hs);
    v16h a1 = frag_a(&As[(wm * 32 + 16 + ln) * 40], hs);
    v16h bbv[4];
#pragma unroll
    for (int tn = 0; tn < 4; ++tn)
      bbv[tn] = frag_b(&Bs[(wn * 64 + tn * 16 + ln) * 40], hs);
#pragma unroll
    for (int tn = 0; tn < 4; ++tn) {
      acc[0][tn] = wmma16(a0, bbv[tn], acc[0][tn]);
      acc[1][tn] = wmma16(a1, bbv[tn], acc[1][tn]);
    }
    __syncthreads();
  }

#pragma unroll
  for (int tm = 0; tm < 2; ++tm)
#pragma unroll
    for (int tn = 0; tn < 4; ++tn) {
      int n = n0 + wn * 64 + tn * 16 + ln;
      float bv = bias[n];
#pragma unroll
      for (int r = 0; r < 8; ++r) {
        int row = m0 + wm * 32 + tm * 16 + r + 8 * hs;
        out[(size_t)row * 1024 + n] = acc[tm][tn][r] + bv;
      }
    }
}

extern "C" void kernel_launch(void* const* d_in, const int* in_sizes, int n_in,
                              void* d_out, int out_size, void* d_ws,
                              size_t ws_size, hipStream_t stream) {
  const float* query = (const float*)d_in[0];
  const float* keyv  = (const float*)d_in[1];
  const float* Wq    = (const float*)d_in[2];
  const float* Wk    = (const float*)d_in[3];
  const float* Wv    = (const float*)d_in[4];
  const float* Wo    = (const float*)d_in[5];
  const float* bo    = (const float*)d_in[6];
  float* out = (float*)d_out;

  const size_t elems = (size_t)BATCH * H * S1 * DK;  // 16,777,216 per tensor
  half_t* qws  = (half_t*)d_ws;
  half_t* kws  = qws + elems;
  half_t* vtws = kws + elems;
  half_t* aws  = vtws + elems;   // 4 x 32 MB f16 = 128 MB total

  dim3 gg(8, 128), bb(256);
  proj_kernel<<<gg, bb, 0, stream>>>(query, Wq, qws, 0);
  proj_kernel<<<gg, bb, 0, stream>>>(keyv,  Wk, kws, 1);
  proj_kernel<<<gg, bb, 0, stream>>>(keyv,  Wv, vtws, 2);
  attn_kernel<<<dim3(32, H, BATCH), dim3(128), 0, stream>>>(qws, kws, vtws, aws);
  outproj_kernel<<<gg, bb, 0, stream>>>(aws, Wo, bo, out);
}
